// LSTM_63737314673186
// MI455X (gfx1250) — matrix operand order repaired and checked by reading, not compiled, not run on hardware
//
#include <hip/hip_runtime.h>

typedef __attribute__((ext_vector_type(16))) __bf16        v16bf;
typedef __attribute__((ext_vector_type(2)))  __bf16        v2bf;
typedef __attribute__((ext_vector_type(8)))  float         v8f;
typedef __attribute__((ext_vector_type(8)))  unsigned int  v8u;

namespace {
constexpr int kB   = 16384;   // batch
constexpr int kH   = 1024;    // hidden
constexpr int kK   = 2048;    // D_IN + H
constexpr int MT   = 128;     // block M tile
constexpr int NTg  = 32;      // block N tile per gate
constexpr int KT   = 32;      // K tile = one WMMA step
constexpr int LDA  = 40;      // LDS row stride (ushorts), 20 banks -> conflict free
constexpr int LDB  = 40;
constexpr int ABUF = MT * LDA;        // 5120 ushorts
constexpr int BBUF = 4 * NTg * LDB;   // 5120 ushorts
constexpr int BUFSZ = ABUF + BBUF;    // 10240 ushorts = 20 KB per buffer
constexpr int NKT  = kK / KT;         // 64 K-steps
}

// pack two f32 -> bf16x2: native cvt_pk if the builtin exists, else
// round-bias + one v_perm_b32 (3 VALU ops per pair, no b16 sub-reg packing)
__device__ __forceinline__ unsigned int pk2bf(float lo, float hi) {
#if __has_builtin(__builtin_amdgcn_cvt_pk_bf16_f32)
  v2bf p = __builtin_amdgcn_cvt_pk_bf16_f32(lo, hi);
  return __builtin_bit_cast(unsigned int, p);
#else
  unsigned int ulo = __float_as_uint(lo) + 0x8000u;   // round-to-nearest bias
  unsigned int uhi = __float_as_uint(hi) + 0x8000u;
  // dst bytes {uhi[3],uhi[2],ulo[3],ulo[2]} -> packed bf16x2 in one v_perm_b32
  return __builtin_amdgcn_perm(uhi, ulo, 0x07060302u);
#endif
}

__device__ __forceinline__ float fast_tanh(float x) {
#if __has_builtin(__builtin_amdgcn_tanhf)
  return __builtin_amdgcn_tanhf(x);            // v_tanh_f32
#elif __has_builtin(__builtin_amdgcn_tanh_f32)
  return __builtin_amdgcn_tanh_f32(x);
#else
  return 2.0f / (1.0f + __expf(-2.0f * x)) - 1.0f;
#endif
}
__device__ __forceinline__ float fast_sig(float x) {
  return __builtin_fmaf(0.5f, fast_tanh(0.5f * x), 0.5f);
}

__device__ __forceinline__ float f4c(const float4& v, int i) {
  return i == 0 ? v.x : i == 1 ? v.y : i == 2 ? v.z : v.w;
}

__global__ void __launch_bounds__(256)
lstm_fused_wmma(const float* __restrict__ X,  const float* __restrict__ Ain,
                const float* __restrict__ Cin,
                const float* __restrict__ Wf, const float* __restrict__ Bf,
                const float* __restrict__ Wu, const float* __restrict__ Bu,
                const float* __restrict__ Wo, const float* __restrict__ Bo,
                const float* __restrict__ Wc, const float* __restrict__ Bc,
                float* __restrict__ OutA0, float* __restrict__ OutA1,
                float* __restrict__ OutC)
{
  __shared__ unsigned short sh[2 * BUFSZ];   // 40 KB, double buffered

  const int tid = threadIdx.x;
  const int m0  = blockIdx.x * MT;
  const int n0  = blockIdx.y * NTg;

  // ---- A global->LDS mapping: 2 threads per row, 16 k each ----
  const int ar = tid >> 1;               // A row 0..127
  const int ak = (tid & 1) << 4;         // A k-segment {0,16}

  // ---- B global->LDS mapping: per gate, 64 threads cover 32x32 as 4k x 4n blocks ----
  const int bg  = tid >> 6;              // gate 0..3
  const int bt  = tid & 63;
  const int btx = bt & 7;                // n block  -> n = btx*4 .. +3
  const int bty = bt >> 3;               // k block  -> k = bty*4 .. +3
  const float* Wg = (bg == 0) ? Wf : (bg == 1) ? Wu : (bg == 2) ? Wo : Wc;

  // ---- wave / fragment mapping ----
  const int wave = tid >> 5;
  const int lane = tid & 31;
  const int wm   = wave >> 1;            // 0..3 : 32-row slice
  const int wn   = wave & 1;             // 0..1 : 16-col slice per gate
  const int half = lane >> 4;
  const int l15  = lane & 15;

  // A fragment K pairs: lanes 0-15 hold K=0..15, lanes 16-31 K=16..31 (contiguous, matches B)
  const int kOffA = half * 16;
  const int kbA[8] = {kOffA, kOffA + 2, kOffA + 4, kOffA + 6,
                      kOffA + 8, kOffA + 10, kOffA + 12, kOffA + 14};
  // B fragment: lanes 0-15 hold K=0..15, lanes 16-31 K=16..31 (contiguous pairs)
  const int kOffB = half * 16;

  v8f acc[4][2];
#pragma unroll
  for (int g = 0; g < 4; ++g)
#pragma unroll
    for (int s = 0; s < 2; ++s)
#pragma unroll
      for (int e = 0; e < 8; ++e) acc[g][s][e] = 0.0f;

  float4 ra[4], rb[4];

  auto loadG = [&](int kt) {
    const int k0 = kt * KT;
    const float* asrc = (k0 < 1024)
        ? X   + (size_t)(m0 + ar) * 1024 + (size_t)(k0 + ak)
        : Ain + (size_t)(m0 + ar) * 1024 + (size_t)(k0 - 1024 + ak);
#pragma unroll
    for (int i = 0; i < 4; ++i) ra[i] = ((const float4*)asrc)[i];
    const float* bsrc = Wg + (size_t)(k0 + bty * 4) * kH + (size_t)(n0 + btx * 4);
#pragma unroll
    for (int j = 0; j < 4; ++j)
      rb[j] = *(const float4*)(bsrc + (size_t)j * kH);   // row k0+bty*4+j, 4 n's
  };

  auto storeL = [&](int buf) {
    unsigned short* base = sh + buf * BUFSZ;
    // A tile: row-major, K-contiguous -> 8 packed cvt + two 16B LDS stores
    unsigned int w[8];
#pragma unroll
    for (int i = 0; i < 4; ++i) {
      w[2 * i]     = pk2bf(ra[i].x, ra[i].y);
      w[2 * i + 1] = pk2bf(ra[i].z, ra[i].w);
    }
    uint4* ap = (uint4*)(base + ar * LDA + ak);
    ap[0] = make_uint4(w[0], w[1], w[2], w[3]);
    ap[1] = make_uint4(w[4], w[5], w[6], w[7]);
    // B tile transposed (N-major): per n, pack 4 k's -> one b64 store (4 total)
    unsigned short* bp = base + ABUF + (bg * NTg + btx * 4) * LDB + bty * 4;
#pragma unroll
    for (int nn = 0; nn < 4; ++nn) {
      uint2 q;
      q.x = pk2bf(f4c(rb[0], nn), f4c(rb[1], nn));
      q.y = pk2bf(f4c(rb[2], nn), f4c(rb[3], nn));
      *(uint2*)(bp + nn * LDB) = q;
    }
  };

  auto compute = [&](int buf) {
    const unsigned short* base = sh + buf * BUFSZ;
    v16bf afrag[2];
#pragma unroll
    for (int s = 0; s < 2; ++s) {
      const unsigned short* ap = base + (wm * 32 + s * 16 + l15) * LDA;
      v8u au;
#pragma unroll
      for (int p = 0; p < 8; ++p)
        au[p] = *(const unsigned int*)(ap + kbA[p]);
      afrag[s] = __builtin_bit_cast(v16bf, au);
    }
#pragma unroll
    for (int g = 0; g < 4; ++g) {
      const unsigned short* bp2 = base + ABUF + (g * NTg + wn * 16 + l15) * LDB;
      v8u bu8;
#pragma unroll
      for (int p = 0; p < 8; ++p)
        bu8[p] = *(const unsigned int*)(bp2 + kOffB + 2 * p);
      v16bf bfrag = __builtin_bit_cast(v16bf, bu8);
#pragma unroll
      for (int s = 0; s < 2; ++s)
        acc[g][s] = __builtin_amdgcn_wmma_f32_16x16x32_bf16(
            false, afrag[s], false, bfrag, (short)0, acc[g][s], false, false);
    }
  };

  // ---- pipelined main loop: single barrier per K-step ----
  loadG(0);
  storeL(0);
#pragma unroll 1
  for (int kt = 0; kt < NKT; ++kt) {
    __syncthreads();                       // buf[kt&1] ready; prev readers of buf[(kt+1)&1] done
    if (kt + 1 < NKT) loadG(kt + 1);       // global loads overlap this tile's compute
    compute(kt & 1);
    if (kt + 1 < NKT) storeL((kt + 1) & 1);
  }

  // ---- fused LSTM epilogue (z never hits memory) ----
  const int gn = n0 + wn * 16 + l15;
  const float bfv = Bf[gn], buv = Bu[gn], bov = Bo[gn], bcv = Bc[gn];
#pragma unroll
  for (int s = 0; s < 2; ++s) {
    const int rbase = m0 + wm * 32 + s * 16 + half * 8;
#pragma unroll
    for (int v = 0; v < 8; ++v) {
      const size_t idx = (size_t)(rbase + v) * kH + gn;
      const float zf = acc[0][s][v] + bfv;
      const float zu = acc[1][s][v] + buv;
      const float zo = acc[2][s][v] + bov;
      const float zc = acc[3][s][v] + bcv;
      const float gf = fast_sig(zf);
      const float gu = fast_sig(zu);
      const float go = fast_sig(zo);
      const float ct = fast_tanh(zc);
      const float co = __builtin_nontemporal_load(Cin + idx);
      const float nc = gu * ct + gf * co;
      const float na = go * fast_tanh(nc);
      __builtin_nontemporal_store(na, OutA0 + idx);
      __builtin_nontemporal_store(na, OutA1 + idx);
      __builtin_nontemporal_store(nc, OutC + idx);
    }
  }
}

extern "C" void kernel_launch(void* const* d_in, const int* in_sizes, int n_in,
                              void* d_out, int out_size, void* d_ws, size_t ws_size,
                              hipStream_t stream) {
  const float* X  = (const float*)d_in[0];
  const float* A  = (const float*)d_in[1];
  const float* C  = (const float*)d_in[2];
  const float* Wf = (const float*)d_in[3];
  const float* Bf = (const float*)d_in[4];
  const float* Wu = (const float*)d_in[5];
  const float* Bu = (const float*)d_in[6];
  const float* Wo = (const float*)d_in[7];
  const float* Bo = (const float*)d_in[8];
  const float* Wc = (const float*)d_in[9];
  const float* Bc = (const float*)d_in[10];
  float* o0 = (float*)d_out;
  float* o1 = o0 + (size_t)kB * kH;
  float* o2 = o1 + (size_t)kB * kH;
  dim3 grid(kB / MT, kH / NTg);
  lstm_fused_wmma<<<grid, dim3(256), 0, stream>>>(X, A, C, Wf, Bf, Wu, Bu,
                                                  Wo, Bo, Wc, Bc, o0, o1, o2);
}
